// BCCLayer_82970178224288
// MI455X (gfx1250) — compile-verified, matrix operand order checked by reading, not run on hardware
//
#include <hip/hip_runtime.h>
#include <hip/hip_bf16.h>

// Problem constants (match reference)
#define LSEQ  2000
#define HDIM  256
#define KDIM  512
#define BATCH 4

typedef __attribute__((ext_vector_type(16))) __bf16 v16bf;
typedef __attribute__((ext_vector_type(8)))  float  v8f;

// Pointer types matching the async global->LDS builtin's parameters:
// param0: int4 in addrspace(1), param1: int4 in addrspace(3).
typedef int v4i __attribute__((ext_vector_type(4)));
typedef __attribute__((address_space(1))) v4i* as1_v4i_p;
typedef __attribute__((address_space(3))) v4i* as3_v4i_p;

#if __has_builtin(__builtin_amdgcn_global_load_async_to_lds_b128)
#define HAVE_ASYNC_LDS 1
#endif

union BF16Frag {
    v16bf v;
    unsigned short u[16];
    uint4 q[2];
};

__device__ inline unsigned short f2bf(float f) {
    unsigned int u = __float_as_uint(f);
    u += 0x7FFFu + ((u >> 16) & 1u);          // round-to-nearest-even
    return (unsigned short)(u >> 16);
}
__device__ inline float bf2f(unsigned short s) {
    return __uint_as_float(((unsigned int)s) << 16);
}

// Wait for all of this wave's outstanding async global->LDS transfers.
__device__ inline void wait_async0() {
#if defined(HAVE_ASYNC_LDS)
#if __has_builtin(__builtin_amdgcn_s_wait_asynccnt)
    __builtin_amdgcn_s_wait_asynccnt(0);
#else
    asm volatile("s_wait_asynccnt 0x0" ::: "memory");
#endif
#endif
}

// Stage a 16-row x KDIM bf16 tile (16 KB) from global into LDS, block-wide.
// Uses the CDNA5 async global->LDS DMA path when the toolchain exposes it.
__device__ inline void stage16rows(const unsigned short* __restrict__ g,
                                   unsigned short* l) {
#if defined(HAVE_ASYNC_LDS)
    for (int i = threadIdx.x; i < 16 * KDIM / 8; i += 128) {
        as1_v4i_p gp = (as1_v4i_p)(uintptr_t)(g + (size_t)i * 8);
        as3_v4i_p lp = (as3_v4i_p)(unsigned int)(uintptr_t)(l + (size_t)i * 8);
        __builtin_amdgcn_global_load_async_to_lds_b128(gp, lp, 0, 0);
    }
#else
    const uint4* src = (const uint4*)g;
    uint4* dst = (uint4*)l;
    for (int i = threadIdx.x; i < 16 * KDIM / 8; i += 128) dst[i] = src[i];
#endif
}

// ---------------------------------------------------------------------------
// 1) Weight-norm scales: scales[i] = g_i / ||V_i||_F
// ---------------------------------------------------------------------------
__global__ __launch_bounds__(256) void norm_kernel(const float* __restrict__ Qv,
                                                   const float* __restrict__ Kv,
                                                   const float* __restrict__ Qg,
                                                   const float* __restrict__ Kg,
                                                   float* __restrict__ scales) {
    const float* V = (blockIdx.x == 0) ? Qv : Kv;
    const float* g = (blockIdx.x == 0) ? Qg : Kg;
    __shared__ float red[256];
    float s = 0.f;
    for (int i = threadIdx.x; i < KDIM * HDIM; i += 256) { float v = V[i]; s += v * v; }
    red[threadIdx.x] = s;
    __syncthreads();
    for (int off = 128; off > 0; off >>= 1) {
        if ((int)threadIdx.x < off) red[threadIdx.x] += red[threadIdx.x + off];
        __syncthreads();
    }
    if (threadIdx.x == 0) scales[blockIdx.x] = g[0] * rsqrtf(red[0]);
}

// ---------------------------------------------------------------------------
// 1b) One-time f32 -> bf16 conversion (optionally scaled by scales[scaleIdx]).
//     Each thread converts 8 elements and stores one 16-byte vector.
// ---------------------------------------------------------------------------
__global__ __launch_bounds__(256) void cvt_kernel(const float* __restrict__ src,
                                                  const float* __restrict__ scales, int scaleIdx,
                                                  unsigned short* __restrict__ dst, int n8) {
    const int i = blockIdx.x * 256 + threadIdx.x;   // index of an 8-element group
    if (i >= n8) return;
    const float s = (scaleIdx >= 0) ? scales[scaleIdx] : 1.0f;
    union { uint4 q; unsigned short u[8]; } o;
    const float4 a = ((const float4*)src)[2 * i];
    const float4 b = ((const float4*)src)[2 * i + 1];
    o.u[0] = f2bf(a.x * s); o.u[1] = f2bf(a.y * s); o.u[2] = f2bf(a.z * s); o.u[3] = f2bf(a.w * s);
    o.u[4] = f2bf(b.x * s); o.u[5] = f2bf(b.y * s); o.u[6] = f2bf(b.z * s); o.u[7] = f2bf(b.w * s);
    ((uint4*)dst)[i] = o.q;
}

// ---------------------------------------------------------------------------
// 2) FC: out = relu(Zb @ Wb^T + bias) [* h_mat]   (all-bf16 GEMM, f32 acc)
//    M = B*L = 8000, N = KDIM = 512, K = HDIM = 256.
//    One wave: 16 N-cols, B frags in 64 VGPRs, sweeps 4 M-tiles.
// ---------------------------------------------------------------------------
__global__ __launch_bounds__(128) void fc_kernel(const unsigned short* __restrict__ Zb,
                                                 const unsigned short* __restrict__ Wb,
                                                 const float* __restrict__ bias,
                                                 const float* __restrict__ hmat, int useH,
                                                 unsigned short* __restrict__ out) {
    const int lane = threadIdx.x & 31;
    const int wave = threadIdx.x >> 5;
    const int n0 = (blockIdx.y * 4 + wave) * 16;
    const int nrow = n0 + (lane & 15);       // B: one N-col per lane (= row of Wb)
    const int koff = (lane < 16) ? 0 : 8;    // 16-bit A/B fragment K split

    // Preload this wave's B fragments (16 cols x 256 K) into registers
    BF16Frag barr[8];
    const unsigned short* bp0 = Wb + (size_t)nrow * HDIM + koff;
    #pragma unroll
    for (int j = 0; j < 8; ++j) {
        barr[j].q[0] = *(const uint4*)(bp0 + j * 32);
        barr[j].q[1] = *(const uint4*)(bp0 + j * 32 + 16);
    }
    const float bv = bias[nrow];
    const float hv = useH ? hmat[nrow] : 1.0f;

    for (int mt = 0; mt < 4; ++mt) {
        const int m0 = (blockIdx.x * 4 + mt) * 16;
        const unsigned short* ap0 = Zb + (size_t)(m0 + (lane & 15)) * HDIM + koff;
        v8f acc = {};
        #pragma unroll
        for (int j = 0; j < 8; ++j) {
            BF16Frag a;
            a.q[0] = *(const uint4*)(ap0 + j * 32);
            a.q[1] = *(const uint4*)(ap0 + j * 32 + 16);
            acc = __builtin_amdgcn_wmma_f32_16x16x32_bf16(false, a.v, false, barr[j].v,
                                                          (short)0, acc, false, false);
        }
        #pragma unroll
        for (int r = 0; r < 8; ++r) {
            const int row = m0 + ((lane < 16) ? r : 8 + r);   // C/D: M split by lane half
            float o = acc[r] + bv;
            o = (o > 0.f) ? o : 0.f;                          // relu BEFORE h (h may be <0)
            out[(size_t)row * KDIM + nrow] = f2bf(o * hv);
        }
    }
}

// ---------------------------------------------------------------------------
// 3) Streaming score GEMM + online column softmax statistics.
//    Score[v,q] = sum_k Arows[v,k] * Bcols[q,k]   (h_bias dropped: shift-inv.)
//    softmax over rows v, per column q; output
//      sOut[q] = colM[q] * (sum_v rowW[v] e^{S[v,q]-M_q}) / (sum_v e^{S[v,q]-M_q})
//    Block = 4 waves, each wave owns 16 columns.
//    B tile (16 cols x 512 K bf16) held in 128 VGPRs for the whole kernel;
//    A row-tiles double-buffered in LDS via async global->LDS DMA.
// ---------------------------------------------------------------------------
__global__ __launch_bounds__(128) void att_kernel(const unsigned short* __restrict__ Arows,
                                                  const unsigned short* __restrict__ Bcols,
                                                  const float* __restrict__ rowW,
                                                  const float* __restrict__ colM,
                                                  float* __restrict__ sOut) {
    const int b    = blockIdx.y;
    const int lane = threadIdx.x & 31;
    const int wave = threadIdx.x >> 5;
    const int NT   = LSEQ / 16;                     // 125 row tiles
    const int colTile = blockIdx.x * 4 + wave;
    const bool valid = colTile < NT;
    const int q0 = (valid ? colTile : NT - 1) * 16; // clamp so preloads stay in-bounds

    __shared__ __align__(16) unsigned short As[2][16 * KDIM];  // 2 x 16 KB double buffer

    const unsigned short* Ab = Arows + (size_t)b * LSEQ * KDIM;
    const unsigned short* Bb = Bcols + (size_t)b * LSEQ * KDIM;

    const int n    = lane & 15;
    const int koff = (lane < 16) ? 0 : 8;
    const int roff = (lane < 16) ? 0 : 8;   // which half of the 16 tile rows this lane owns

    // ---- Preload this wave's B fragments into registers (constant over v) ----
    BF16Frag barr[16];
    const unsigned short* bp0 = Bb + (size_t)(q0 + n) * KDIM + koff;
    #pragma unroll
    for (int j = 0; j < 16; ++j) {
        barr[j].q[0] = *(const uint4*)(bp0 + j * 32);
        barr[j].q[1] = *(const uint4*)(bp0 + j * 32 + 16);
    }

    // ---- Prime the pipeline: stage tile 0 ----
    stage16rows(Ab, As[0]);
    wait_async0();
    __syncthreads();

    float lm = -3.0e38f, ls = 0.f, lt = 0.f;   // running max / sum-exp / weighted sum-exp

    for (int vt = 0; vt < NT; ++vt) {
        const int cur = vt & 1;
        // Issue async DMA for the next tile into the other buffer
        if (vt + 1 < NT)
            stage16rows(Ab + (size_t)(vt + 1) * 16 * KDIM, As[cur ^ 1]);

        // Row-weight mask for this lane's 8 rows: two float4 loads (contiguous)
        const float4* rp = (const float4*)(rowW + b * LSEQ + vt * 16 + roff);
        const float4 rw0 = rp[0];
        const float4 rw1 = rp[1];
        const float w[8] = { rw0.x, rw0.y, rw0.z, rw0.w, rw1.x, rw1.y, rw1.z, rw1.w };

        // Compute 16x16 score tile: A frags from LDS, B frags from registers
        v8f acc = {};
        const unsigned short* Asc = As[cur];
        #pragma unroll
        for (int j = 0; j < 16; ++j) {
            BF16Frag a;
            a.q[0] = *(const uint4*)&Asc[(lane & 15) * KDIM + j * 32 + koff];
            a.q[1] = *(const uint4*)&Asc[(lane & 15) * KDIM + j * 32 + koff + 16];
            acc = __builtin_amdgcn_wmma_f32_16x16x32_bf16(false, a.v, false, barr[j].v,
                                                          (short)0, acc, false, false);
        }

        // Online update: this lane holds 8 rows of column q0+n
        float tm = acc[0];
        #pragma unroll
        for (int r = 1; r < 8; ++r) tm = fmaxf(tm, acc[r]);
        const float nm = fmaxf(lm, tm);
        const float rescale = __expf(lm - nm);
        ls *= rescale; lt *= rescale;
        #pragma unroll
        for (int r = 0; r < 8; ++r) {
            const float e = __expf(acc[r] - nm);
            ls += e;
            lt += w[r] * e;
        }
        lm = nm;

        wait_async0();      // next buffer's DMA complete (no-op on last iteration)
        __syncthreads();    // all waves done reading As[cur] before it is overwritten
    }

    // Column q0+n lives in lanes n and n+16 -> merge across the half-wave
    const float om = __shfl_xor(lm, 16, 32);
    const float os = __shfl_xor(ls, 16, 32);
    const float ot = __shfl_xor(lt, 16, 32);
    const float M = fmaxf(lm, om);
    const float S = ls * __expf(lm - M) + os * __expf(om - M);
    const float T = lt * __expf(lm - M) + ot * __expf(om - M);

    if (valid && lane < 16) {
        const int q = q0 + lane;
        sOut[b * LSEQ + q] = colM[b * LSEQ + q] * (T / S);
    }
}

// ---------------------------------------------------------------------------
// 4) pooled parts: pooled[term,b,k] = sum_i s[b,i] * Mat[b,i,k]
// ---------------------------------------------------------------------------
__global__ __launch_bounds__(512) void pool_kernel(const float* __restrict__ s1,
                                                   const float* __restrict__ s2,
                                                   const unsigned short* __restrict__ YK,
                                                   const unsigned short* __restrict__ XK,
                                                   float* __restrict__ pooled) {
    const int b = blockIdx.x;
    const int term = blockIdx.y;
    const int k = threadIdx.x;
    const float* sv = (term == 0) ? s1 : s2;
    const unsigned short* mat = (term == 0) ? YK : XK;
    float acc = 0.f;
    for (int i = 0; i < LSEQ; ++i) {
        if (i + 8 < LSEQ)
            __builtin_prefetch(&mat[(size_t)(b * LSEQ + i + 8) * KDIM + k], 0, 1);
        const float s = sv[b * LSEQ + i];
        acc += s * bf2f(mat[(size_t)(b * LSEQ + i) * KDIM + k]);
    }
    pooled[(term * BATCH + b) * KDIM + k] = acc;
}

// ---------------------------------------------------------------------------
// 5) cross-batch layernorm of pooled means
// ---------------------------------------------------------------------------
__global__ __launch_bounds__(512) void final_kernel(const float* __restrict__ pooled,
                                                    const float* __restrict__ gamma,
                                                    const float* __restrict__ beta,
                                                    float* __restrict__ out) {
    const int k = threadIdx.x;
    float p[BATCH];
    float mu = 0.f;
    #pragma unroll
    for (int b = 0; b < BATCH; ++b) {
        p[b] = (pooled[b * KDIM + k] + pooled[(BATCH + b) * KDIM + k]) * (1.0f / LSEQ);
        mu += p[b];
    }
    mu *= (1.0f / BATCH);
    float var = 0.f;
    #pragma unroll
    for (int b = 0; b < BATCH; ++b) { const float d = p[b] - mu; var += d * d; }
    var *= (1.0f / BATCH);
    const float inv = rsqrtf(var + 1e-5f);
    #pragma unroll
    for (int b = 0; b < BATCH; ++b)
        out[b * KDIM + k] = gamma[k] * (p[b] - mu) * inv + beta[k];
}

// ---------------------------------------------------------------------------
extern "C" void kernel_launch(void* const* d_in, const int* in_sizes, int n_in,
                              void* d_out, int out_size, void* d_ws, size_t ws_size,
                              hipStream_t stream) {
    (void)in_sizes; (void)n_in; (void)out_size; (void)ws_size;
    const float* X     = (const float*)d_in[0];
    const float* Y     = (const float*)d_in[1];
    const float* mask1 = (const float*)d_in[2];
    const float* mask2 = (const float*)d_in[3];
    const float* Qv    = (const float*)d_in[4];
    const float* Qg    = (const float*)d_in[5];
    const float* Qb    = (const float*)d_in[6];
    const float* Kv    = (const float*)d_in[7];
    const float* Kg    = (const float*)d_in[8];
    const float* Kb    = (const float*)d_in[9];
    const float* h_mat = (const float*)d_in[10];
    // d_in[11] = h_bias: unused — softmax is shift-invariant, the constant cancels.
    const float* gamma = (const float*)d_in[12];
    const float* beta  = (const float*)d_in[13];
    float* out = (float*)d_out;

    char* ws = (char*)d_ws;
    size_t off = 0;
    float* scales = (float*)(ws + off); off += 256;
    const size_t actBytes = (size_t)BATCH * LSEQ * KDIM * sizeof(unsigned short); // 8 MB each
    const size_t inBytes  = (size_t)BATCH * LSEQ * HDIM * sizeof(unsigned short); // 4 MB each
    const size_t wBytes   = (size_t)KDIM * HDIM * sizeof(unsigned short);         // 256 KB each
    unsigned short* XQh = (unsigned short*)(ws + off); off += actBytes; // relu(X Wq^T+qb)*h
    unsigned short* YK  = (unsigned short*)(ws + off); off += actBytes; // relu(Y Wk^T+kb)
    unsigned short* YQh = (unsigned short*)(ws + off); off += actBytes; // relu(Y Wq^T+qb)*h
    unsigned short* XK  = (unsigned short*)(ws + off); off += actBytes; // relu(X Wk^T+kb)
    unsigned short* Xbf = (unsigned short*)(ws + off); off += inBytes;  // bf16(X)
    unsigned short* Ybf = (unsigned short*)(ws + off); off += inBytes;  // bf16(Y)
    unsigned short* Wq  = (unsigned short*)(ws + off); off += wBytes;   // bf16(scaleQ*Qv)
    unsigned short* Wk  = (unsigned short*)(ws + off); off += wBytes;   // bf16(scaleK*Kv)
    float* s1     = (float*)(ws + off); off += (size_t)BATCH * LSEQ * sizeof(float);
    float* s2     = (float*)(ws + off); off += (size_t)BATCH * LSEQ * sizeof(float);
    float* pooled = (float*)(ws + off); off += (size_t)2 * BATCH * KDIM * sizeof(float);

    norm_kernel<<<2, 256, 0, stream>>>(Qv, Kv, Qg, Kg, scales);

    // One-time conversions to bf16 (weights fused with the weight-norm scale)
    const int actN8 = BATCH * LSEQ * HDIM / 8;   // 256000
    const int wN8   = KDIM * HDIM / 8;           // 16384
    cvt_kernel<<<(actN8 + 255) / 256, 256, 0, stream>>>(X,  scales, -1, Xbf, actN8);
    cvt_kernel<<<(actN8 + 255) / 256, 256, 0, stream>>>(Y,  scales, -1, Ybf, actN8);
    cvt_kernel<<<(wN8 + 255) / 256, 256, 0, stream>>>(Qv, scales,  0, Wq,  wN8);
    cvt_kernel<<<(wN8 + 255) / 256, 256, 0, stream>>>(Kv, scales,  1, Wk,  wN8);

    dim3 fcGrid(BATCH * LSEQ / 64, KDIM / 64);   // (125, 8), 4 waves/block, 4 M-tiles/wave
    fc_kernel<<<fcGrid, 128, 0, stream>>>(Xbf, Wq, Qb, h_mat, 1, XQh);
    fc_kernel<<<fcGrid, 128, 0, stream>>>(Ybf, Wk, Kb, h_mat, 0, YK);
    fc_kernel<<<fcGrid, 128, 0, stream>>>(Ybf, Wq, Qb, h_mat, 1, YQh);
    fc_kernel<<<fcGrid, 128, 0, stream>>>(Xbf, Wk, Kb, h_mat, 0, XK);

    dim3 attGrid((LSEQ / 16 + 3) / 4, BATCH);    // (32, 4)
    // A1: rows v=XQh, cols q=YK, row weights mask1, column mask mask2
    att_kernel<<<attGrid, 128, 0, stream>>>(XQh, YK, mask1, mask2, s1);
    // A2: rows q=YQh, cols v=XK, row weights mask2, column mask mask1
    att_kernel<<<attGrid, 128, 0, stream>>>(YQh, XK, mask2, mask1, s2);

    pool_kernel<<<dim3(BATCH, 2), 512, 0, stream>>>(s1, s2, YK, XK, pooled);
    final_kernel<<<1, 512, 0, stream>>>(pooled, gamma, beta, out);
}
